// Decoder_49100066128290
// MI455X (gfx1250) — matrix operand order repaired and checked by reading, not compiled, not run on hardware
//
#include <hip/hip_runtime.h>
#include <cmath>

typedef __attribute__((ext_vector_type(16))) __bf16        v16bf;
typedef __attribute__((ext_vector_type(8)))  float         v8f;
typedef __attribute__((ext_vector_type(4)))  unsigned int  u32x4;
typedef unsigned short u16;

union Frag { v16bf v; u32x4 q[2]; };

__device__ __forceinline__ u16 f2bfu(float f) {
    unsigned int u = __float_as_uint(f);
    unsigned int r = (u + 0x7FFFu + ((u >> 16) & 1u)) >> 16;
    return (u16)r;
}

__device__ __forceinline__ v8f zero8() {
    v8f z;
#pragma unroll
    for (int i = 0; i < 8; ++i) z[i] = 0.0f;
    return z;
}

__device__ __forceinline__ v8f wmma_bf16(v16bf a, v16bf b, v8f c) {
    return __builtin_amdgcn_wmma_f32_16x16x32_bf16(
        false, a, false, b, (short)0, c, false, false);
}

// CDNA5 async global->LDS copy, GV addressing (CDNA5 ISA).
__device__ __forceinline__ void async_ld_b128(unsigned lds_off, const void* gptr) {
    asm volatile("global_load_async_to_lds_b128 %0, %1, off"
                 :: "v"(lds_off), "v"((unsigned long long)gptr)
                 : "memory");
}
__device__ __forceinline__ void wait_async_5() {
    asm volatile("s_wait_asynccnt 5" ::: "memory");
}
__device__ __forceinline__ void wait_async_0() {
    asm volatile("s_wait_asynccnt 0" ::: "memory");
}

// ---------------- f32 -> bf16 convert ----------------
__global__ void cvt_bf16_kernel(const float* __restrict__ in, u16* __restrict__ out, int n) {
    int i = blockIdx.x * blockDim.x + threadIdx.x;
    int stride = gridDim.x * blockDim.x;
    for (; i < n; i += stride) out[i] = f2bfu(in[i]);
}

// ---------------- embedding gather -> bf16 (4096 x 512) ----------------
__global__ void gather_emb_kernel(const float* __restrict__ embed, const int* __restrict__ target,
                                  u16* __restrict__ out, int n) {
    int i = blockIdx.x * blockDim.x + threadIdx.x;
    int stride = gridDim.x * blockDim.x;
    for (; i < n; i += stride) {
        int row = i >> 9;
        int e   = i & 511;
        int t   = target[row];
        out[i]  = f2bfu(embed[(size_t)t * 512 + e]);
    }
}

// ---------------- ctx projections: ses_inf (16x1024), h0 (16x1024) ----------------
__global__ void ctx_kernel(const float* __restrict__ ctx, const float* __restrict__ Wses,
                           const float* __restrict__ Ws2d, const float* __restrict__ bs2d,
                           float* __restrict__ ses_inf, float* __restrict__ h0) {
    int id = blockIdx.x * blockDim.x + threadIdx.x;
    int b  = id >> 11;
    int j  = id & 2047;
    const float* c = ctx + b * 1024;
    if (j < 1024) {
        const float* w = Wses + (size_t)j * 1024;
        float s = 0.f;
        for (int k = 0; k < 1024; ++k) s += c[k] * w[k];
        ses_inf[b * 1024 + j] = s;
    } else {
        int n = j - 1024;
        const float* w = Ws2d + (size_t)n * 1024;
        float s = 0.f;
        for (int k = 0; k < 1024; ++k) s += c[k] * w[k];
        h0[b * 1024 + n] = tanhf(s + bs2d[n]);
    }
}

// ---------------- stage one 64xK32 A tile + 256xK32 B tile into LDS (async) ----------------
// 5 async b128 instructions per thread (1 A + 4 B) => 5 per wave for ASYNCcnt accounting.
__device__ __forceinline__ void stage_tiles(const u16* __restrict__ A, const u16* __restrict__ W,
                                            int K, int mblk, int nblk, int k0,
                                            unsigned shA_off, unsigned shB_off, int tid)
{
    int rowA = tid >> 2, ca = (tid & 3) * 8;
    async_ld_b128(shA_off + tid * 16,
                  (const void*)(A + (size_t)(mblk + rowA) * K + k0 + ca));
#pragma unroll
    for (int c = 0; c < 4; ++c) {
        int cc = tid + c * 256;
        int rowB = cc >> 2, cb = (cc & 3) * 8;
        async_ld_b128(shB_off + cc * 16,
                      (const void*)(W + (size_t)(nblk + rowB) * K + k0 + cb));
    }
}

// one k-step of WMMA work from a resolved LDS buffer
__device__ __forceinline__ void gemm_compute(const u16* __restrict__ shAbuf,
                                             const u16* __restrict__ shBbuf,
                                             v8f acc[8], int wr, int wc, int ml, int hi)
{
    Frag a;
    const u16* ap = shAbuf + (wr * 16 + ml) * 32 + hi * 8;
    a.q[0] = *(const u32x4*)(ap);
    a.q[1] = *(const u32x4*)(ap + 16);

    Frag b[8];
#pragma unroll
    for (int t = 0; t < 8; ++t) {
        const u16* bp = shBbuf + (wc * 128 + t * 16 + ml) * 32 + hi * 8;
        b[t].q[0] = *(const u32x4*)(bp);
        b[t].q[1] = *(const u32x4*)(bp + 16);
    }
#pragma unroll
    for (int t = 0; t < 8; ++t)
        acc[t] = wmma_bf16(a.v, b[t].v, acc[t]);
}

// ---------------- LDS-staged, async double-buffered bf16 WMMA GEMM ----------------
// Block: 256 threads (8 waves) computing 64M x 256N; wave tile 16M x 128N.
// Main loop manually unrolled 2x so the LDS buffer index is compile-time constant.
// Requires K % 64 == 0 (here K is 512 or 1024).
__global__ __launch_bounds__(256) void gemm_bf16_kernel(
    const u16* __restrict__ A,      // M x K bf16 row-major
    const u16* __restrict__ W,      // N x K bf16 row-major (B = W^T)
    const float* __restrict__ bias, // [N] or null
    const float* __restrict__ addMN,// [M x N] or null
    const float* __restrict__ addBN,// [(M>>8) x N] or null
    float* __restrict__ D,          // M x N f32
    int K, int N, int nt_store)
{
    __shared__ __align__(16) u16 shA[2][64 * 32];    //  8 KB
    __shared__ __align__(16) u16 shB[2][256 * 32];   // 32 KB

    const int tid  = threadIdx.x;
    const int lane = tid & 31;
    const int wv   = tid >> 5;
    const int wr   = wv & 3;       // 0..3 : M sub-row
    const int wc   = wv >> 2;      // 0..1 : N half
    const int ml   = lane & 15;
    const int hi   = lane >> 4;
    const int mblk = blockIdx.x * 64;
    const int nblk = blockIdx.y * 256;

    unsigned aoff[2], boff[2];
#pragma unroll
    for (int i = 0; i < 2; ++i) {
        aoff[i] = (unsigned)(unsigned long long)(void*)(&shA[i][0]);
        boff[i] = (unsigned)(unsigned long long)(void*)(&shB[i][0]);
    }

    v8f acc[8];
#pragma unroll
    for (int t = 0; t < 8; ++t) acc[t] = zero8();

    stage_tiles(A, W, K, mblk, nblk, 0, aoff[0], boff[0], tid);

    for (int kk = 0; kk < K; kk += 64) {
        // ---- phase A: compute buf0 (kk), prefetch kk+32 into buf1 ----
        stage_tiles(A, W, K, mblk, nblk, kk + 32, aoff[1], boff[1], tid);
        wait_async_5();                 // buf0's 5 async ops done (in-order)
        __syncthreads();
        gemm_compute(&shA[0][0], &shB[0][0], acc, wr, wc, ml, hi);
        __syncthreads();

        // ---- phase B: compute buf1 (kk+32), prefetch kk+64 into buf0 ----
        if (kk + 64 < K) {
            stage_tiles(A, W, K, mblk, nblk, kk + 64, aoff[0], boff[0], tid);
            wait_async_5();
        } else {
            wait_async_0();
        }
        __syncthreads();
        gemm_compute(&shA[1][0], &shB[1][0], acc, wr, wc, ml, hi);
        __syncthreads();
    }

#pragma unroll
    for (int t = 0; t < 8; ++t) {
        int n = nblk + wc * 128 + t * 16 + ml;
        float bv = bias ? bias[n] : 0.0f;
#pragma unroll
        for (int r = 0; r < 8; ++r) {
            int m = mblk + wr * 16 + r + 8 * hi;
            float v = acc[t][r] + bv;
            if (addMN) v += addMN[(size_t)m * N + n];
            if (addBN) v += addBN[(size_t)(m >> 8) * N + n];
            float* p = &D[(size_t)m * N + n];
            if (nt_store) __builtin_nontemporal_store(v, p);   // write-once 524MB output
            else          *p = v;
        }
    }
}

// ---------------- GRU scan: one workgroup, h resident in LDS ----------------
__global__ __launch_bounds__(1024) void gru_kernel(
    const u16*   __restrict__ Whh,   // 3072 x 1024 bf16
    const float* __restrict__ bhh,   // 3072
    const float* __restrict__ xproj, // (16*256) x 3072 f32
    const float* __restrict__ h0,    // 16 x 1024 f32
    u16*         __restrict__ hs_bf, // (16*256) x 1024 bf16
    int S)
{
    __shared__ float sh_h[16 * 1024];                 // 64 KB
    __shared__ __align__(16) u16 sh_hbf[16 * 1024];   // 32 KB
    const int tid  = threadIdx.x;
    const int lane = tid & 31;
    const int w    = tid >> 5;
    const int ml   = lane & 15;
    const int hi   = lane >> 4;

    for (int i = tid; i < 16 * 1024; i += 1024) sh_h[i] = h0[i];

    for (int s = 0; s < S; ++s) {
        __syncthreads();
        for (int i = tid; i < 16 * 1024; i += 1024) sh_hbf[i] = f2bfu(sh_h[i]);
        __syncthreads();

        v8f acc[6];
#pragma unroll
        for (int q = 0; q < 6; ++q) acc[q] = zero8();

        const u16* arow = sh_hbf + ml * 1024 + hi * 8;
#pragma unroll
        for (int k0 = 0; k0 < 1024; k0 += 32) {
            Frag a;
            a.q[0] = *(const u32x4*)(arow + k0);
            a.q[1] = *(const u32x4*)(arow + k0 + 16);
            Frag b[6];
#pragma unroll
            for (int q = 0; q < 6; ++q) {
                int g = q >> 1, t = q & 1;
                int n0 = g * 1024 + w * 32 + t * 16;
                const u16* bp = Whh + (size_t)(n0 + ml) * 1024 + k0 + hi * 8;
                b[q].q[0] = *(const u32x4*)(bp);
                b[q].q[1] = *(const u32x4*)(bp + 16);
            }
#pragma unroll
            for (int q = 0; q < 6; ++q)
                acc[q] = wmma_bf16(a.v, b[q].v, acc[q]);
        }

#pragma unroll
        for (int t = 0; t < 2; ++t) {
            int j = w * 32 + t * 16 + ml;
            float bh_r = bhh[j];
            float bh_z = bhh[1024 + j];
            float bh_n = bhh[2048 + j];
#pragma unroll
            for (int r = 0; r < 8; ++r) {
                int b = r + 8 * hi;
                float hr = acc[0 + t][r];
                float hz = acc[2 + t][r];
                float hn = acc[4 + t][r];
                size_t xbase = (size_t)(b * 256 + s) * 3072;
                float xr = xproj[xbase + j];
                float xz = xproj[xbase + 1024 + j];
                float xn = xproj[xbase + 2048 + j];
                float rg = 1.0f / (1.0f + expf(-(xr + hr + bh_r)));
                float zg = 1.0f / (1.0f + expf(-(xz + hz + bh_z)));
                float ng = tanhf(xn + rg * (hn + bh_n));
                float hold = sh_h[b * 1024 + j];
                float hnew = (1.0f - zg) * ng + zg * hold;
                sh_h[b * 1024 + j] = hnew;
                hs_bf[(size_t)(b * 256 + s) * 1024 + j] = f2bfu(hnew);
            }
        }
    }
}

// ---------------- maxout over adjacent pairs + bf16 ----------------
__global__ void maxout_kernel(const float* __restrict__ total, u16* __restrict__ mx_bf, int n) {
    int i = blockIdx.x * blockDim.x + threadIdx.x;
    int stride = gridDim.x * blockDim.x;
    for (; i < n; i += stride) {
        int m = i >> 9;
        int j = i & 511;
        float a = total[(size_t)m * 1024 + 2 * j];
        float b = total[(size_t)m * 1024 + 2 * j + 1];
        mx_bf[i] = f2bfu(fmaxf(a, b));
    }
}

// ---------------- workspace layout (bytes) ----------------
#define OFF_EMBED_BF   (size_t)0            // 32000*512*2 = 32,768,000
#define OFF_WIH_BF     (size_t)32768000     // 3072*512*2
#define OFF_WHH_BF     (size_t)35913728     // 3072*1024*2
#define OFF_WEMB_BF    (size_t)42205184     // 1024*512*2
#define OFF_WDEC_BF    (size_t)43253760     // 1024*1024*2
#define OFF_EMB_BF     (size_t)45350912     // 4096*512*2 (reused as mx_bf)
#define OFF_SES        (size_t)49545216     // 16*1024*4
#define OFF_H0         (size_t)49610752     // 16*1024*4
#define OFF_EMBINF     (size_t)49676288     // 4096*1024*4
#define OFF_XPROJ      (size_t)66453504     // 4096*3072*4 (reused as total)
#define OFF_HSBF       (size_t)116785152    // 4096*1024*2
// total ws use: 125,173,760 bytes

extern "C" void kernel_launch(void* const* d_in, const int* in_sizes, int n_in,
                              void* d_out, int out_size, void* d_ws, size_t ws_size,
                              hipStream_t stream) {
    const float* ctx    = (const float*)d_in[0];
    const int*   target = (const int*)  d_in[1];
    const float* embed  = (const float*)d_in[2];
    const float* W_s2d  = (const float*)d_in[3];
    const float* b_s2d  = (const float*)d_in[4];
    const float* W_dec  = (const float*)d_in[5];
    const float* W_ses  = (const float*)d_in[6];
    const float* W_emb  = (const float*)d_in[7];
    const float* b_emb  = (const float*)d_in[8];
    const float* W_ih   = (const float*)d_in[9];
    const float* W_hh   = (const float*)d_in[10];
    const float* b_ih   = (const float*)d_in[11];
    const float* b_hh   = (const float*)d_in[12];
    float* out = (float*)d_out;

    char* ws = (char*)d_ws;
    u16*   embed_bf = (u16*)(ws + OFF_EMBED_BF);
    u16*   Wih_bf   = (u16*)(ws + OFF_WIH_BF);
    u16*   Whh_bf   = (u16*)(ws + OFF_WHH_BF);
    u16*   Wemb_bf  = (u16*)(ws + OFF_WEMB_BF);
    u16*   Wdec_bf  = (u16*)(ws + OFF_WDEC_BF);
    u16*   emb_bf   = (u16*)(ws + OFF_EMB_BF);
    float* ses_inf  = (float*)(ws + OFF_SES);
    float* h0       = (float*)(ws + OFF_H0);
    float* emb_inf  = (float*)(ws + OFF_EMBINF);
    float* x_proj   = (float*)(ws + OFF_XPROJ);
    float* total    = (float*)(ws + OFF_XPROJ);
    u16*   mx_bf    = (u16*)(ws + OFF_EMB_BF);
    u16*   hs_bf    = (u16*)(ws + OFF_HSBF);

    {
        int n;
        n = 32000 * 512;  cvt_bf16_kernel<<<(n + 1023) / 1024, 256, 0, stream>>>(embed, embed_bf, n);
        n = 3072 * 512;   cvt_bf16_kernel<<<(n + 1023) / 1024, 256, 0, stream>>>(W_ih, Wih_bf, n);
        n = 3072 * 1024;  cvt_bf16_kernel<<<(n + 1023) / 1024, 256, 0, stream>>>(W_hh, Whh_bf, n);
        n = 1024 * 512;   cvt_bf16_kernel<<<(n + 1023) / 1024, 256, 0, stream>>>(W_emb, Wemb_bf, n);
        n = 1024 * 1024;  cvt_bf16_kernel<<<(n + 1023) / 1024, 256, 0, stream>>>(W_dec, Wdec_bf, n);
    }

    {
        int n = 4096 * 512;
        gather_emb_kernel<<<(n + 255) / 256, 256, 0, stream>>>(embed, target, emb_bf, n);
    }

    ctx_kernel<<<128, 256, 0, stream>>>(ctx, W_ses, W_s2d, b_s2d, ses_inf, h0);

    // emb_inf = emb @ W_emb^T + b_emb   (M=4096, K=512, N=1024)
    gemm_bf16_kernel<<<dim3(64, 4), 256, 0, stream>>>(
        emb_bf, Wemb_bf, b_emb, nullptr, nullptr, emb_inf, 512, 1024, 0);

    // x_proj = emb @ W_ih^T + b_ih      (M=4096, K=512, N=3072)
    gemm_bf16_kernel<<<dim3(64, 12), 256, 0, stream>>>(
        emb_bf, Wih_bf, b_ih, nullptr, nullptr, x_proj, 512, 3072, 0);

    // GRU scan (sequential recurrence): single workgroup, 32 waves
    gru_kernel<<<1, 1024, 0, stream>>>(Whh_bf, b_hh, x_proj, h0, hs_bf, 256);

    // total = hs @ W_dec^T + ses_inf[b] + emb_inf   (M=4096, K=1024, N=1024)
    gemm_bf16_kernel<<<dim3(64, 4), 256, 0, stream>>>(
        hs_bf, Wdec_bf, nullptr, emb_inf, ses_inf, total, 1024, 1024, 0);

    {
        int n = 4096 * 512;
        maxout_kernel<<<(n + 255) / 256, 256, 0, stream>>>(total, mx_bf, n);
    }

    // logits = mx @ embed^T  (M=4096, K=512, N=32000) -> d_out, NT stores
    gemm_bf16_kernel<<<dim3(64, 125), 256, 0, stream>>>(
        mx_bf, embed_bf, nullptr, nullptr, nullptr, out, 512, 32000, 1);
}
